// MinGPT_58557584113675
// MI455X (gfx1250) — matrix operand order repaired and checked by reading, not compiled, not run on hardware
//
#include <hip/hip_runtime.h>

typedef __attribute__((ext_vector_type(16))) __bf16 v16bf;
typedef __attribute__((ext_vector_type(8)))  __bf16 v8bf;
typedef __attribute__((ext_vector_type(8)))  float  v8f;

// 16-byte chunk type for the gfx1250 async Global->LDS builtin
typedef int b128vec __attribute__((vector_size(16)));
#define GLOBAL_AS __attribute__((address_space(1)))
#define LDS_AS    __attribute__((address_space(3)))

#if defined(__HIP_DEVICE_COMPILE__) && \
    __has_builtin(__builtin_amdgcn_global_load_async_to_lds_b128) && \
    __has_builtin(__builtin_amdgcn_s_wait_asynccnt)
#define USE_ASYNC_LDS 1
#else
#define USE_ASYNC_LDS 0
#endif

#define NEMB 128
#define TLEN 256
#define NLAYER 6
#define FFDIM 512
#define MROWS 16384      // 64 * 256
#define VOCAB 3149
#define VOCAB_PAD 3200   // pad to multiple of 64 (4-tile macro columns)

// ---------------------------------------------------------------------------
// Embedding: x[b,t,c] = tok_emb[tokens[b,t], c] + pos_emb[t, c]
// ---------------------------------------------------------------------------
__global__ __launch_bounds__(256) void embed_kernel(
    const int* __restrict__ tokens, const float* __restrict__ tok_emb,
    const float* __restrict__ pos_emb, float* __restrict__ x)
{
    long i = (long)blockIdx.x * 256 + threadIdx.x;   // < MROWS*NEMB
    int c   = (int)(i & (NEMB - 1));
    long r  = i >> 7;                                // row = b*256 + t
    int t   = (int)(r & (TLEN - 1));
    int tok = tokens[r];
    x[i] = tok_emb[(long)tok * NEMB + c] + pos_emb[t * NEMB + c];
}

// ---------------------------------------------------------------------------
// LayerNorm over C=128 -> bf16 output (one row per 128-thread block)
// ---------------------------------------------------------------------------
__global__ __launch_bounds__(128) void layernorm_kernel(
    const float* __restrict__ x, const float* __restrict__ g,
    const float* __restrict__ b, __bf16* __restrict__ out)
{
    __shared__ float red[128];
    int r = blockIdx.x, c = threadIdx.x;
    float v = x[(long)r * NEMB + c];
    red[c] = v; __syncthreads();
    for (int off = 64; off > 0; off >>= 1) {
        if (c < off) red[c] += red[c + off];
        __syncthreads();
    }
    float mu = red[0] * (1.0f / 128.0f);
    __syncthreads();
    float d = v - mu;
    red[c] = d * d; __syncthreads();
    for (int off = 64; off > 0; off >>= 1) {
        if (c < off) red[c] += red[c + off];
        __syncthreads();
    }
    float var = red[0] * (1.0f / 128.0f);
    float rs = rsqrtf(var + 1e-5f);
    out[(long)r * NEMB + c] = (__bf16)(d * rs * g[c] + b[c]);
}

// ---------------------------------------------------------------------------
// Weight repack: per-head (H,C,D) f32 -> bf16 WT[n=h*16+d][k=c]
// ---------------------------------------------------------------------------
__global__ __launch_bounds__(256) void repack_headw(
    const float* __restrict__ src, __bf16* __restrict__ dst)
{
    int i = blockIdx.x * 256 + threadIdx.x;  // < 128*128
    int c = i & 127, n = i >> 7;
    int h = n >> 4, d = n & 15;
    dst[n * 128 + c] = (__bf16)src[h * 2048 + c * 16 + d];
}

// ---------------------------------------------------------------------------
// Weight repack: (K,N) f32 row-major -> bf16 WT[Npad][K] (transpose, zero-pad)
// ---------------------------------------------------------------------------
__global__ __launch_bounds__(256) void transpose_pack(
    const float* __restrict__ src, __bf16* __restrict__ dst,
    int K, int N, int Npad)
{
    long i = (long)blockIdx.x * 256 + threadIdx.x;
    if (i >= (long)Npad * K) return;
    int  k = (int)(i % K);
    long n = i / K;
    dst[i] = (n < N) ? (__bf16)src[(long)k * N + n] : (__bf16)0.0f;
}

// ---------------------------------------------------------------------------
// bf16 WMMA GEMM with macro-tiling:
//   block = 128 rows x 64 cols (8 waves x 16 rows, 4 N-tiles per wave).
//   64xK B panel staged once per block into LDS via async Global->LDS
//   (gfx1250 GLOBAL_LOAD_ASYNC_TO_LDS_B128, drained with s_wait_asynccnt),
//   shared by all 8 waves. A fragment loaded once per k-step, 4 WMMAs each.
// ---------------------------------------------------------------------------
__global__ __launch_bounds__(256) void gemm_bf16_wmma(
    const __bf16* __restrict__ A, const __bf16* __restrict__ Bt,
    const float* __restrict__ bias, const float* __restrict__ residual,
    float* __restrict__ outF, __bf16* __restrict__ outH,
    int M, int N, int Nb4, int K, int relu)
{
    extern __shared__ __bf16 ldsB[];                 // [64][K]
    int lane = threadIdx.x & 31;
    int wave = threadIdx.x >> 5;
    int nb = blockIdx.x % Nb4;
    int mb = blockIdx.x / Nb4;

    // ---- cooperative stage of the 64xK bf16 B panel into LDS ----
    {
        const __bf16* Bbase = Bt + (long)nb * 64 * K;
        int chunks = (64 * K) / 8;                   // 16B chunks
        for (int c = threadIdx.x; c < chunks; c += 256) {
#if USE_ASYNC_LDS
            __builtin_amdgcn_global_load_async_to_lds_b128(
                (GLOBAL_AS b128vec*)(b128vec*)(void*)(__bf16*)(Bbase + (long)c * 8),
                (LDS_AS    b128vec*)(b128vec*)(void*)(ldsB + (long)c * 8),
                0, 0);
#else
            *(v8bf*)(ldsB + (long)c * 8) = *(const v8bf*)(Bbase + (long)c * 8);
#endif
        }
#if USE_ASYNC_LDS
        __builtin_amdgcn_s_wait_asynccnt(0);
#endif
        __syncthreads();
    }

    int ka = (lane < 16) ? 0 : 8;                    // ISA 16-bit A layout
    int kb = (lane < 16) ? 0 : 8;                    // ISA 16-bit B layout
    int arow = (mb * 8 + wave) * 16 + (lane & 15);
    const __bf16* Ap  = A + (long)arow * K + ka;
    const __bf16* Bl0 = ldsB + (lane & 15) * K + kb;

    v8f acc[4] = {{}, {}, {}, {}};
    for (int k0 = 0; k0 < K; k0 += 32) {
        if (k0 + 32 < K) __builtin_prefetch(Ap + k0 + 32, 0, 3);
        v8bf a0 = *(const v8bf*)(Ap + k0);           // K: ka+0..7
        v8bf a1 = *(const v8bf*)(Ap + k0 + 16);      // K: ka+16..23
        v16bf a;
        #pragma unroll
        for (int i = 0; i < 8; ++i) { a[i] = a0[i]; a[i + 8] = a1[i]; }
        #pragma unroll
        for (int j = 0; j < 4; ++j) {
            const __bf16* bp = Bl0 + j * 16 * K + k0;
            v8bf b0 = *(const v8bf*)bp;              // K: kb+0..7
            v8bf b1 = *(const v8bf*)(bp + 16);       // K: kb+16..23
            v16bf bfr;
            #pragma unroll
            for (int i = 0; i < 8; ++i) { bfr[i] = b0[i]; bfr[i + 8] = b1[i]; }
            acc[j] = __builtin_amdgcn_wmma_f32_16x16x32_bf16(
                false, a, false, bfr, (short)0, acc[j], false, false);
        }
    }

    int mbase = (mb * 8 + wave) * 16 + ((lane < 16) ? 0 : 8);
    #pragma unroll
    for (int j = 0; j < 4; ++j) {
        int bcol = (nb * 4 + j) * 16 + (lane & 15);
        if (bcol < N) {
            float bv = bias ? bias[bcol] : 0.0f;
            #pragma unroll
            for (int i = 0; i < 8; ++i) {
                int m = mbase + i;
                float v = acc[j][i] + bv;
                if (relu) v = v > 0.0f ? v : 0.0f;
                long idx = (long)m * N + bcol;
                if (residual) v += residual[idx];
                if (outF) outF[idx] = v;
                if (outH) outH[idx] = (__bf16)v;
            }
        }
    }
}

// ---------------------------------------------------------------------------
// Fused causal attention, one workgroup per (b,h).
// LDS: scores f32 [256][256] (256KB) + v^T bf16 [16][256] (8KB) = 270336 B.
// qkv layout: [m=b*256+t][384] bf16, q at col h*16, k at 128+h*16, v at 256+h*16
// ---------------------------------------------------------------------------
__global__ __launch_bounds__(256) void attention_kernel(
    const __bf16* __restrict__ qkv, __bf16* __restrict__ attn_out)
{
    extern __shared__ float smem[];
    float*  sc = smem;                               // [256][256]
    __bf16* vT = (__bf16*)(smem + TLEN * TLEN);      // [16][256]

    int bh = blockIdx.x;            // 0..511
    int b = bh >> 3, h = bh & 7;
    int tid  = threadIdx.x;
    int lane = tid & 31, wave = tid >> 5;
    const __bf16* base = qkv + (long)b * TLEN * 384;
    const float scale = 0.088388347648318447f;       // 128^-0.5 (note: C, not D)

    // stage v^T : vT[d][t] = v[t][d]
    {
        const __bf16* vrow = base + (long)tid * 384 + 256 + h * 16;
        #pragma unroll
        for (int d = 0; d < 16; ++d) vT[d * TLEN + tid] = vrow[d];
    }

    // phase 1: scores = q . k^T (D=16 padded to K=32 with zeros), lower tiles only
    int ka = (lane < 16) ? 0 : 8;
    {
        int idx = -1;
        for (int ti = 0; ti < 16; ++ti) {
            for (int si = 0; si <= ti; ++si) {
                ++idx;
                if ((idx & 7) != wave) continue;     // wave-uniform
                int t = ti * 16 + (lane & 15);
                int s = si * 16 + (lane & 15);
                v16bf a, kfr;
                const __bf16* qrow = base + (long)t * 384 + h * 16 + ka;
                v8bf qv = *(const v8bf*)qrow;        // d = ka..ka+7
                #pragma unroll
                for (int i = 0; i < 8; ++i) { a[i] = qv[i]; a[i + 8] = (__bf16)0.0f; }
                {                                    // d = ka..ka+7, K=16..31 -> zero pad
                    const __bf16* krow = base + (long)s * 384 + 128 + h * 16 + ka;
                    v8bf k0 = *(const v8bf*)krow;
                    #pragma unroll
                    for (int i = 0; i < 8; ++i) { kfr[i] = k0[i]; kfr[i + 8] = (__bf16)0.0f; }
                }
                v8f acc = {};
                acc = __builtin_amdgcn_wmma_f32_16x16x32_bf16(
                    false, a, false, kfr, (short)0, acc, false, false);
                int mbr = ti * 16 + ((lane < 16) ? 0 : 8);
                int n   = si * 16 + (lane & 15);
                #pragma unroll
                for (int i = 0; i < 8; ++i)
                    sc[(mbr + i) * TLEN + n] = acc[i] * scale;
            }
        }
    }
    __syncthreads();

    // phase 2: causal softmax, one row per thread; zero s>t
    {
        int t = tid;
        float* row = sc + t * TLEN;
        float mx = row[0];
        for (int s = 1; s <= t; ++s) mx = fmaxf(mx, row[s]);
        float sum = 0.0f;
        for (int s = 0; s <= t; ++s) { float e = __expf(row[s] - mx); row[s] = e; sum += e; }
        float inv = 1.0f / sum;
        for (int s = 0; s <= t; ++s) row[s] *= inv;
        for (int s = t + 1; s < TLEN; ++s) row[s] = 0.0f;
    }
    __syncthreads();

    // phase 3: attn = wei . v  (A = wei f32->bf16 from LDS, B = v^T bf16 from LDS)
    for (int ti = wave; ti < 16; ti += 8) {          // wave-uniform
        int t = ti * 16 + (lane & 15);
        int d = lane & 15;
        v8f acc = {};
        int kmax = ti * 16 + 16;
        for (int k0 = 0; k0 < kmax; k0 += 32) {
            const float* ar = sc + t * TLEN + k0 + ka;
            v16bf a, bfr;
            #pragma unroll
            for (int i = 0; i < 8; ++i) {
                a[i]     = (__bf16)ar[i];            // s = k0+ka+0..7
                a[i + 8] = (__bf16)ar[16 + i];       // s = k0+ka+16..23
            }
            const __bf16* br = vT + d * TLEN + k0 + ((lane < 16) ? 0 : 8);
            v8bf b0 = *(const v8bf*)br;
            v8bf b1 = *(const v8bf*)(br + 16);
            #pragma unroll
            for (int i = 0; i < 8; ++i) { bfr[i] = b0[i]; bfr[i + 8] = b1[i]; }
            acc = __builtin_amdgcn_wmma_f32_16x16x32_bf16(
                false, a, false, bfr, (short)0, acc, false, false);
        }
        int mbr = ti * 16 + ((lane < 16) ? 0 : 8);
        #pragma unroll
        for (int i = 0; i < 8; ++i)
            attn_out[((long)(b * TLEN + mbr + i)) * NEMB + h * 16 + d] = (__bf16)acc[i];
    }
}

// ---------------------------------------------------------------------------
// Host: orchestrate the full forward pass on `stream`
// ---------------------------------------------------------------------------
extern "C" void kernel_launch(void* const* d_in, const int* in_sizes, int n_in,
                              void* d_out, int out_size, void* d_ws, size_t ws_size,
                              hipStream_t stream) {
    const int*   tokens  = (const int*)  d_in[0];
    const float* tok_emb = (const float*)d_in[1];
    const float* pos_emb = (const float*)d_in[2];
    const float* Wq      = (const float*)d_in[3];
    const float* Wk      = (const float*)d_in[4];
    const float* Wv      = (const float*)d_in[5];
    const float* Wproj   = (const float*)d_in[6];
    const float* bproj   = (const float*)d_in[7];
    const float* ln1_g   = (const float*)d_in[8];
    const float* ln1_b   = (const float*)d_in[9];
    const float* ln2_g   = (const float*)d_in[10];
    const float* ln2_b   = (const float*)d_in[11];
    const float* W1      = (const float*)d_in[12];
    const float* b1      = (const float*)d_in[13];
    const float* W2      = (const float*)d_in[14];
    const float* b2      = (const float*)d_in[15];
    const float* lnf_g   = (const float*)d_in[16];
    const float* lnf_b   = (const float*)d_in[17];
    const float* Wlm     = (const float*)d_in[18];
    const float* blm     = (const float*)d_in[19];
    float* logits = (float*)d_out;

    // workspace carve-up (256B aligned)
    char* p = (char*)d_ws;
    auto alloc = [&](size_t bytes) -> char* {
        char* r = p; p += (bytes + 255) & ~(size_t)255; return r;
    };
    float*  x     = (float*) alloc((size_t)MROWS * NEMB * 4);
    __bf16* xn    = (__bf16*)alloc((size_t)MROWS * NEMB * 2);
    __bf16* qkv   = (__bf16*)alloc((size_t)MROWS * 384 * 2);
    __bf16* attnb = (__bf16*)alloc((size_t)MROWS * NEMB * 2);
    __bf16* hbuf  = (__bf16*)alloc((size_t)MROWS * FFDIM * 2);
    __bf16* wqkv  = (__bf16*)alloc((size_t)NLAYER * 384 * NEMB * 2);
    __bf16* wpr   = (__bf16*)alloc((size_t)NLAYER * NEMB * NEMB * 2);
    __bf16* w1t   = (__bf16*)alloc((size_t)NLAYER * FFDIM * NEMB * 2);
    __bf16* w2t   = (__bf16*)alloc((size_t)NLAYER * NEMB * FFDIM * 2);
    __bf16* wlmt  = (__bf16*)alloc((size_t)VOCAB_PAD * NEMB * 2);

    // --- weight repack (bf16, N-major / K-contiguous) ---
    for (int l = 0; l < NLAYER; ++l) {
        const long hw = (long)l * 8 * 128 * 16;
        repack_headw<<<64, 256, 0, stream>>>(Wq + hw, wqkv + (long)l * 384 * 128 + 0 * 128 * 128);
        repack_headw<<<64, 256, 0, stream>>>(Wk + hw, wqkv + (long)l * 384 * 128 + 1 * 128 * 128);
        repack_headw<<<64, 256, 0, stream>>>(Wv + hw, wqkv + (long)l * 384 * 128 + 2 * 128 * 128);
        transpose_pack<<<64,   256, 0, stream>>>(Wproj + (long)l * 128 * 128, wpr + (long)l * 128 * 128, 128, 128, 128);
        transpose_pack<<<256,  256, 0, stream>>>(W1    + (long)l * 128 * 512, w1t + (long)l * 512 * 128, 128, 512, 512);
        transpose_pack<<<256,  256, 0, stream>>>(W2    + (long)l * 512 * 128, w2t + (long)l * 128 * 512, 512, 128, 128);
    }
    transpose_pack<<<1600, 256, 0, stream>>>(Wlm, wlmt, 128, VOCAB, VOCAB_PAD);

    // --- embeddings ---
    embed_kernel<<<(MROWS * NEMB) / 256, 256, 0, stream>>>(tokens, tok_emb, pos_emb, x);

    const int attn_lds   = TLEN * TLEN * 4 + 16 * TLEN * 2; // 270336 B < 320KB WGP LDS
    const int panel128   = 64 * 128 * 2;                    // 16 KB B panel (K=128)
    const int panel512   = 64 * 512 * 2;                    // 64 KB B panel (K=512)
    const int MB = MROWS / 128;                             // 128 macro row-blocks

    // --- transformer blocks ---
    for (int l = 0; l < NLAYER; ++l) {
        layernorm_kernel<<<MROWS, 128, 0, stream>>>(x, ln1_g + l * 128, ln1_b + l * 128, xn);

        // fused QKV: (16384,128) x (128,384) -> bf16   (Nb4 = 384/64 = 6)
        gemm_bf16_wmma<<<MB * 6, 256, panel128, stream>>>(
            xn, wqkv + (long)l * 384 * 128, nullptr, nullptr, nullptr, qkv,
            MROWS, 384, 6, 128, 0);

        attention_kernel<<<512, 256, attn_lds, stream>>>(qkv, attnb);

        // proj + bias + residual -> x (f32)            (Nb4 = 128/64 = 2)
        gemm_bf16_wmma<<<MB * 2, 256, panel128, stream>>>(
            attnb, wpr + (long)l * 128 * 128, bproj + l * 128, x, x, nullptr,
            MROWS, 128, 2, 128, 0);

        layernorm_kernel<<<MROWS, 128, 0, stream>>>(x, ln2_g + l * 128, ln2_b + l * 128, xn);

        // FFN1 + bias + relu -> bf16                   (Nb4 = 512/64 = 8)
        gemm_bf16_wmma<<<MB * 8, 256, panel128, stream>>>(
            xn, w1t + (long)l * 512 * 128, b1 + l * 512, nullptr, nullptr, hbuf,
            MROWS, 512, 8, 128, 1);

        // FFN2 + bias + residual -> x (f32)            (Nb4 = 2, K=512)
        gemm_bf16_wmma<<<MB * 2, 256, panel512, stream>>>(
            hbuf, w2t + (long)l * 128 * 512, b2 + l * 128, x, x, nullptr,
            MROWS, 128, 2, 512, 0);
    }

    // --- final LN + LM head ---
    layernorm_kernel<<<MROWS, 128, 0, stream>>>(x, lnf_g, lnf_b, xn);
    gemm_bf16_wmma<<<MB * (VOCAB_PAD / 64), 256, panel128, stream>>>(
        xn, wlmt, blm, nullptr, logits, nullptr,
        MROWS, VOCAB, VOCAB_PAD / 64, 128, 0);
}
